// STAttenAttention_39178691674142
// MI455X (gfx1250) — matrix operand order repaired and checked by reading, not compiled, not run on hardware
//
#include <hip/hip_runtime.h>
#include <hip/hip_bf16.h>

// ---------------------------------------------------------------------------
// STAtten linear attention, MI455X (gfx1250, wave32, WMMA).
//
// Shapes: x[4,4096,1024], W*[1024,1024], bo[1024], H=16, D=64.
// Pipeline (bf16 intermediates, fp32 WMMA accumulate):
//   K0: convert Wq/Wk/Wv/Wo fp32 -> bf16                       (ws)
//   K1: q = x Wq^T  [B,H,N,D]   (computed as Wq x^T so stores pack over d)
//       kT,vT = (x W^T)^T [B,H,D,N] (stores pack over n)
//   K2: kvT[b,h] = (0.125 * kT vT^T)^T  (64x64/head, stores pack over d1)
//   K3: attn = q kv  (computed as kv^T q^T so stores pack over d)
//   K4: out = attn Wo^T + bo  (computed as Wo attn^T: packed v8f stores + bias)
//
// Every operand load is a contiguous v8bf/v16bf/v8f; every epilogue store is
// a packed 16B/32B vector store; `mat` rides blockIdx.y so branches stay SALU.
//
// Workspace (~136.5 MB): [W bf16 8MB][q 32MB][kT 32MB][vT 32MB][kvT 0.5MB][attn 32MB]
// ---------------------------------------------------------------------------

using bf16 = __bf16;
typedef bf16  v16bf __attribute__((ext_vector_type(16)));
typedef bf16  v8bf  __attribute__((ext_vector_type(8)));
typedef float v8f   __attribute__((ext_vector_type(8)));

__device__ __forceinline__ v8f wmma_bf16(v16bf a, v16bf b, v8f c) {
    // (neg_a, A, neg_b, B, c_mod, C, reuse_a, reuse_b)
    return __builtin_amdgcn_wmma_f32_16x16x32_bf16(false, a, false, b,
                                                   (short)0, c, false, false);
}

// A-fragment (16x32): two 8-element bf16 chunks at p and p+8.
__device__ __forceinline__ v16bf frag_bf(const bf16* __restrict__ p0,
                                         const bf16* __restrict__ p1) {
    v8bf lo = *(const v8bf*)p0;
    v8bf hi = *(const v8bf*)p1;
    v16bf a;
#pragma unroll
    for (int i = 0; i < 8; ++i) { a[i] = lo[i]; a[i + 8] = hi[i]; }
    return a;
}

// Fragment from two 8-float chunks (fp32 -> bf16 convert).
__device__ __forceinline__ v16bf frag_f32(const float* __restrict__ p0,
                                          const float* __restrict__ p1) {
    v8f f0 = *(const v8f*)p0;
    v8f f1 = *(const v8f*)p1;
    v16bf a;
#pragma unroll
    for (int i = 0; i < 8; ++i) { a[i] = (bf16)f0[i]; a[i + 8] = (bf16)f1[i]; }
    return a;
}

// Pack 8 fp32 accumulator elements into one 16B bf16 store.
__device__ __forceinline__ void store_pack8(bf16* __restrict__ dst, v8f acc,
                                            float scale) {
    v8bf p;
#pragma unroll
    for (int i = 0; i < 8; ++i) p[i] = (bf16)(acc[i] * scale);
    *(v8bf*)dst = p;
}

// ------------------------- K0: fp32 -> bf16 convert ------------------------
__global__ void __launch_bounds__(256)
cvt_f32_bf16_kernel(const float* __restrict__ src, bf16* __restrict__ dst, int n) {
    for (int i = blockIdx.x * blockDim.x + threadIdx.x; i < n;
         i += gridDim.x * blockDim.x)
        dst[i] = (bf16)src[i];
}

// ------------------------- K1: QKV projections -----------------------------
// grid dim3(1024, 3), 256 thr. blockIdx.y = mat (0=q,1=k,2=v) -> scalar branch.
// Per mat: waves = mPair(512) x strip(16) = 8192. K = 1024.
__global__ void __launch_bounds__(256)
qkv_gemm_kernel(const float* __restrict__ x, const bf16* __restrict__ wbf,
                bf16* __restrict__ qb, bf16* __restrict__ kT,
                bf16* __restrict__ vT) {
    const int mat = blockIdx.y;
    const int waveId = blockIdx.x * 8 + (threadIdx.x >> 5);
    const int lane = threadIdx.x & 31;
    const int l = lane & 15, hi = lane >> 4;
    const int m0 = (waveId & 511) * 32;        // 32 seq rows
    const int h = waveId >> 9;                 // head / 64-col strip
    const int o0 = h * 64;

    if (mat == 0) {
        // q path: y^T = Wq x^T. A = Wq rows (o), B = x rows (n) as columns.
        const bf16* __restrict__ Aw = wbf + (size_t)(o0 + l) * 1024 + hi * 16;
        const float* __restrict__ Bx = x + (size_t)(m0 + l) * 1024 + hi * 16;
        v8f c[4][2] = {};
        for (int k0 = 0; k0 < 1024; k0 += 32) {
            v16bf b0 = frag_f32(Bx + k0, Bx + k0 + 8);
            v16bf b1 = frag_f32(Bx + 16 * 1024 + k0, Bx + 16 * 1024 + k0 + 8);
            __builtin_prefetch(Bx + k0 + 64, 0, 0);       // global_prefetch_b8
            v16bf a[4];
#pragma unroll
            for (int t = 0; t < 4; ++t)
                a[t] = frag_bf(Aw + (size_t)t * 16 * 1024 + k0,
                               Aw + (size_t)t * 16 * 1024 + k0 + 8);
#pragma unroll
            for (int t = 0; t < 4; ++t) {
                c[t][0] = wmma_bf16(a[t], b0, c[t][0]);
                c[t][1] = wmma_bf16(a[t], b1, c[t][1]);
            }
        }
        // lane holds seq row g; e walks d (contiguous) -> packed stores.
#pragma unroll
        for (int mi = 0; mi < 2; ++mi) {
            const int g = m0 + mi * 16 + l;
            const int b2 = g >> 12, n = g & 4095;
            bf16* __restrict__ dst =
                qb + (size_t)((b2 * 16 + h) * 4096 + n) * 64 + hi * 8;
#pragma unroll
            for (int t = 0; t < 4; ++t)
                store_pack8(dst + t * 16, c[t][mi], 1.0f);
        }
    } else {
        // k/v path: y = x W^T. A = x rows (n), B = W rows (o) as columns.
        const bf16* __restrict__ W = wbf + (size_t)mat * (1024 * 1024);
        const float* __restrict__ ar0 = x + (size_t)(m0 + l) * 1024 + hi * 16;
        const float* __restrict__ ar1 = ar0 + 16 * 1024;
        const bf16* __restrict__ Bw = W + (size_t)(o0 + l) * 1024 + hi * 16;
        v8f c[2][4] = {};
        for (int k0 = 0; k0 < 1024; k0 += 32) {
            v16bf a0 = frag_f32(ar0 + k0, ar0 + k0 + 8);
            v16bf a1 = frag_f32(ar1 + k0, ar1 + k0 + 8);
            __builtin_prefetch(ar0 + k0 + 64, 0, 0);
            v16bf b[4];
#pragma unroll
            for (int t = 0; t < 4; ++t)
                b[t] = *(const v16bf*)(Bw + (size_t)t * 16 * 1024 + k0);
#pragma unroll
            for (int t = 0; t < 4; ++t) {
                c[0][t] = wmma_bf16(a0, b[t], c[0][t]);
                c[1][t] = wmma_bf16(a1, b[t], c[1][t]);
            }
        }
        // lane holds d col; e walks n (contiguous in [B,H,D,N]) -> packed.
        bf16* __restrict__ obuf = (mat == 1) ? kT : vT;
#pragma unroll
        for (int mi = 0; mi < 2; ++mi) {
            const int gb = m0 + mi * 16 + hi * 8;      // pack stays in one b
            const int b2 = gb >> 12, nb = gb & 4095;
#pragma unroll
            for (int t = 0; t < 4; ++t) {
                const int d = t * 16 + l;
                store_pack8(obuf + (size_t)((b2 * 16 + h) * 64 + d) * 4096 + nb,
                            c[mi][t], 1.0f);
            }
        }
    }
}

// ------------------------- K2: kv = scale * k^T v --------------------------
// waves: bh(64) x dStrip(4) = 256 ; 32 blocks of 256. K-dim = N = 4096.
__global__ void __launch_bounds__(256)
kv_gemm_kernel(const bf16* __restrict__ kT, const bf16* __restrict__ vT,
               bf16* __restrict__ kvT) {
    const int waveId = blockIdx.x * 8 + (threadIdx.x >> 5);
    const int lane = threadIdx.x & 31;
    const int l = lane & 15, hi = lane >> 4;
    const int bh = waveId >> 2;
    const int m0 = (waveId & 3) * 16;          // d1-rows of this strip

    const bf16* __restrict__ Ab = kT + (size_t)bh * 64 * 4096 + (size_t)(m0 + l) * 4096 + hi * 16;
    const bf16* __restrict__ Bb = vT + (size_t)bh * 64 * 4096 + (size_t)l * 4096 + hi * 16;

    v8f c[4] = {};
    for (int n0 = 0; n0 < 4096; n0 += 32) {
        v16bf a = frag_bf(Ab + n0, Ab + n0 + 8);
        __builtin_prefetch(Ab + n0 + 64, 0, 0);
        v16bf b[4];
#pragma unroll
        for (int t = 0; t < 4; ++t)
            b[t] = *(const v16bf*)(Bb + (size_t)t * 16 * 4096 + n0);
#pragma unroll
        for (int t = 0; t < 4; ++t)
            c[t] = wmma_bf16(a, b[t], c[t]);
    }
    // store transposed, scale = D^-0.5 = 0.125; e walks d1 (contiguous).
#pragma unroll
    for (int t = 0; t < 4; ++t) {
        const int col = t * 16 + l;
        store_pack8(kvT + (size_t)bh * 4096 + col * 64 + m0 + hi * 8,
                    c[t], 0.125f);
    }
}

// ------------------------- K3: attn = q kv ---------------------------------
// Computed as out^T = kv^T q^T: A = kvT rows (d2), B = q rows (n) as columns.
// waves: bh(64) x nTile(256) = 16384 ; 2048 blocks of 256. K = D = 64.
__global__ void __launch_bounds__(256)
av_gemm_kernel(const bf16* __restrict__ qb, const bf16* __restrict__ kvT,
               bf16* __restrict__ attn) {
    const int waveId = blockIdx.x * 8 + (threadIdx.x >> 5);
    const int lane = threadIdx.x & 31;
    const int l = lane & 15, hi = lane >> 4;
    const int bh = waveId >> 8;
    const int m0 = (waveId & 255) * 16;        // 16 seq rows

    const bf16* __restrict__ Ab = kvT + (size_t)bh * 4096 + (size_t)l * 64 + hi * 16;
    const bf16* __restrict__ Bb = qb + (size_t)bh * 4096 * 64 + (size_t)(m0 + l) * 64 + hi * 16;

    v8f c[4] = {};
#pragma unroll
    for (int k0 = 0; k0 < 64; k0 += 32) {
        v16bf b = *(const v16bf*)(Bb + k0);
#pragma unroll
        for (int t = 0; t < 4; ++t) {
            v16bf a = frag_bf(Ab + (size_t)t * 16 * 64 + k0,
                              Ab + (size_t)t * 16 * 64 + k0 + 8);
            c[t] = wmma_bf16(a, b, c[t]);
        }
    }
    // lane holds seq row n; e walks d (contiguous in [B,N,C]) -> packed.
    const int b2 = bh >> 4, hh = bh & 15;
    const int n = m0 + l;
    bf16* __restrict__ dst =
        attn + (size_t)(b2 * 4096 + n) * 1024 + hh * 64 + hi * 8;
#pragma unroll
    for (int t = 0; t < 4; ++t)
        store_pack8(dst + t * 16, c[t], 1.0f);
}

// ------------------------- K4: final projection + bias ---------------------
// Computed as out^T = Wo attn^T: A = Wo rows (o), B = attn rows (m) as cols.
// waves: mPair(512) x oStrip(16) = 8192 ; 1024 blocks of 256. fp32 output.
__global__ void __launch_bounds__(256)
out_gemm_kernel(const bf16* __restrict__ attn, const bf16* __restrict__ Wo,
                const float* __restrict__ bo, float* __restrict__ out) {
    const int waveId = blockIdx.x * 8 + (threadIdx.x >> 5);
    const int lane = threadIdx.x & 31;
    const int l = lane & 15, hi = lane >> 4;
    const int m0 = (waveId & 511) * 32;
    const int o0 = (waveId >> 9) * 64;

    const bf16* __restrict__ Aw = Wo + (size_t)(o0 + l) * 1024 + hi * 16;
    const bf16* __restrict__ Bx = attn + (size_t)(m0 + l) * 1024 + hi * 16;

    v8f c[4][2] = {};
    for (int k0 = 0; k0 < 1024; k0 += 32) {
        v16bf b0 = *(const v16bf*)(Bx + k0);
        v16bf b1 = *(const v16bf*)(Bx + 16 * 1024 + k0);
        __builtin_prefetch(Bx + k0 + 64, 0, 0);
        v16bf a[4];
#pragma unroll
        for (int t = 0; t < 4; ++t)
            a[t] = frag_bf(Aw + (size_t)t * 16 * 1024 + k0,
                           Aw + (size_t)t * 16 * 1024 + k0 + 8);
#pragma unroll
        for (int t = 0; t < 4; ++t) {
            c[t][0] = wmma_bf16(a[t], b0, c[t][0]);
            c[t][1] = wmma_bf16(a[t], b1, c[t][1]);
        }
    }
    // lane holds output row m; e walks oc (contiguous) -> v8f stores + bias.
#pragma unroll
    for (int mi = 0; mi < 2; ++mi) {
        const int m = m0 + mi * 16 + l;
#pragma unroll
        for (int t = 0; t < 4; ++t) {
            const int ocb = o0 + t * 16 + hi * 8;
            v8f bias = *(const v8f*)(bo + ocb);
            v8f r = c[t][mi] + bias;
            *(v8f*)(out + (size_t)m * 1024 + ocb) = r;
        }
    }
}

// ---------------------------------------------------------------------------
extern "C" void kernel_launch(void* const* d_in, const int* in_sizes, int n_in,
                              void* d_out, int out_size, void* d_ws, size_t ws_size,
                              hipStream_t stream) {
    (void)in_sizes; (void)n_in; (void)out_size; (void)ws_size;

    const float* x  = (const float*)d_in[0];
    const float* Ws[4] = { (const float*)d_in[1], (const float*)d_in[2],
                           (const float*)d_in[3], (const float*)d_in[4] };
    const float* bo = (const float*)d_in[5];
    float* out = (float*)d_out;

    bf16* wbf  = (bf16*)d_ws;                                  // 4 x 1M bf16
    bf16* qb   = (bf16*)((char*)d_ws + 4ull * 2 * 1024 * 1024);
    bf16* kT   = qb  + 16777216ull;
    bf16* vT   = kT  + 16777216ull;
    bf16* kvT  = vT  + 16777216ull;
    bf16* attn = kvT + 262144ull;

    for (int i = 0; i < 4; ++i)
        cvt_f32_bf16_kernel<<<256, 256, 0, stream>>>(Ws[i], wbf + (size_t)i * 1048576, 1048576);

    qkv_gemm_kernel<<<dim3(1024, 3), 256, 0, stream>>>(x, wbf, qb, kT, vT);
    kv_gemm_kernel <<<32,   256, 0, stream>>>(kT, vT, kvT);
    av_gemm_kernel <<<2048, 256, 0, stream>>>(qb, kvT, attn);
    out_gemm_kernel<<<1024, 256, 0, stream>>>(attn, wbf + 3ull * 1048576, bo, out);
}
